// DilateAttention_20134806684286
// MI455X (gfx1250) — matrix operand order repaired and checked by reading, not compiled, not run on hardware
//
#include <hip/hip_runtime.h>

// DilateAttention (3x3 neighborhood attention, dilation 1), B=8 C=384 H=W=64,
// head_dim=64, 6 heads. Memory-bound (~200MB @ 23.3TB/s ~= 8.6us floor).
// QK^T done via v_wmma_f32_16x16x32_f16 (diagonal-extraction trick), softmax +
// AV in fp32 VALU, all tiles staged through LDS with coalesced loads.

#define BB 8
#define CC 384
#define HH 64
#define WW 64
#define HD 64
#define NH (CC / HD) /* 6 */
#define SCALE 0.125f /* 64^-0.5 */

typedef _Float16 h8_t  __attribute__((ext_vector_type(8)));
typedef _Float16 v16h  __attribute__((ext_vector_type(16)));
typedef float    v8f   __attribute__((ext_vector_type(8)));

#define SHUF16(a, b) __builtin_shufflevector((a), (b), 0,1,2,3,4,5,6,7,8,9,10,11,12,13,14,15)

// LDS row strides (in elements) chosen so every b128 access is 16B-aligned
// and banks are spread: 72 halfs = 144B, 68 floats = 272B.
#define QS 72
#define KS 72
#define VS 68

__global__ __launch_bounds__(32) void
dilate_attn_kernel(const float* __restrict__ q, const float* __restrict__ k,
                   const float* __restrict__ v, float* __restrict__ out)
{
    __shared__ __align__(16) _Float16 qsm[16 * QS];   // q tile   [x][d] f16
    __shared__ __align__(16) _Float16 ksm[54 * KS];   // k tile   [dy][x0-1..x0+16][d] f16
    __shared__ __align__(16) float    vsm[54 * VS];   // v tile   [dy][x0-1..x0+16][d] f32
    __shared__ __align__(16) float    sc[9 * 16];     // scores -> weights [offset][pixel]

    const int lane = threadIdx.x;          // 0..31 (wave32)
    const int bid  = blockIdx.x;
    const int xt   = bid & 3;
    const int y    = (bid >> 2) & 63;
    const int hh   = (bid >> 8) % NH;
    const int b    = bid / (4 * 64 * NH);
    const int x0   = xt * 16;
    const int c0   = hh * HD;

    const int xl = lane & 15;              // pixel-in-tile / N index
    const int hi = lane >> 4;              // 0 or 1 (lane half)

    // ---------------- stage 0: fill LDS (coalesced along x) ----------------
    // q: 64 channels x 16 x, converted to f16
    #pragma unroll 4
    for (int it = 0; it < 32; ++it) {
        const int d = 2 * it + hi;
        const float val = q[((b * CC + c0 + d) * HH + y) * WW + x0 + xl];
        qsm[xl * QS + d] = (_Float16)val;
    }
    // k main region: 3 rows x 64 channels x 16 x (zero-fill OOB rows)
    #pragma unroll 4
    for (int it = 0; it < 96; ++it) {
        const int dyd = 2 * it + hi;
        const int dy = dyd >> 6, d = dyd & 63;
        const int yy = y + dy - 1;
        const bool ok = (yy >= 0) && (yy < HH);
        const float val = ok ? k[((b * CC + c0 + d) * HH + yy) * WW + x0 + xl] : 0.f;
        ksm[(dy * 18 + 1 + xl) * KS + d] = (_Float16)val;
    }
    // k halo columns x0-1 and x0+16
    #pragma unroll
    for (int it = 0; it < 12; ++it) {
        const int flat = it * 32 + lane;        // 0..383
        const int col  = flat & 1;
        const int rest = flat >> 1;
        const int d = rest & 63, dy = rest >> 6;
        const int yy = y + dy - 1;
        const int xg = x0 + (col ? 16 : -1);
        const bool ok = (yy >= 0) && (yy < HH) && (xg >= 0) && (xg < WW);
        const float val = ok ? k[((b * CC + c0 + d) * HH + yy) * WW + xg] : 0.f;
        ksm[(dy * 18 + (col ? 17 : 0)) * KS + d] = (_Float16)val;
    }
    // v main region (kept fp32)
    #pragma unroll 4
    for (int it = 0; it < 96; ++it) {
        const int dyd = 2 * it + hi;
        const int dy = dyd >> 6, d = dyd & 63;
        const int yy = y + dy - 1;
        const bool ok = (yy >= 0) && (yy < HH);
        const float val = ok ? v[((b * CC + c0 + d) * HH + yy) * WW + x0 + xl] : 0.f;
        vsm[(dy * 18 + 1 + xl) * VS + d] = val;
    }
    // v halo
    #pragma unroll
    for (int it = 0; it < 12; ++it) {
        const int flat = it * 32 + lane;
        const int col  = flat & 1;
        const int rest = flat >> 1;
        const int d = rest & 63, dy = rest >> 6;
        const int yy = y + dy - 1;
        const int xg = x0 + (col ? 16 : -1);
        const bool ok = (yy >= 0) && (yy < HH) && (xg >= 0) && (xg < WW);
        const float val = ok ? v[((b * CC + c0 + d) * HH + yy) * WW + xg] : 0.f;
        vsm[(dy * 18 + (col ? 17 : 0)) * VS + d] = val;
    }
    __syncthreads();

    // ---------------- stage 1: QK^T via WMMA (diagonal extraction) ---------
    // A operand (16-bit A 16x32 layout): lane L row M=L&15;
    //   halfs 0-7  -> K = {0..7 | 8..15}  (+8 for lane>=16)
    //   halfs 8-15 -> K = {16..23 | 24..31}
    const h8_t* qrow = (const h8_t*)(qsm + xl * QS);
    const v16h A0 = SHUF16(qrow[hi],     qrow[2 + hi]);   // d 0..31
    const v16h A1 = SHUF16(qrow[4 + hi], qrow[6 + hi]);   // d 32..63

    #pragma unroll
    for (int o = 0; o < 9; ++o) {
        const int dy = o / 3, dx = o % 3;
        // B operand (16-bit B 32x16): lane L col N=L&15; halfs 0-7 -> K={0..7 | 8..15}, halfs 8-15 -> K={16..23 | 24..31}
        const h8_t* krow = (const h8_t*)(ksm + (dy * 18 + xl + dx) * KS);
        const int hb = hi * 2;
        const v16h B0 = SHUF16(krow[hi],     krow[2 + hi]);   // d 0..31
        const v16h B1 = SHUF16(krow[4 + hi], krow[6 + hi]);   // d 32..63

        v8f acc = {};
        acc = __builtin_amdgcn_wmma_f32_16x16x32_f16(false, A0, false, B0,
                                                     (short)0, acc, false, false);
        acc = __builtin_amdgcn_wmma_f32_16x16x32_f16(false, A1, false, B1,
                                                     (short)0, acc, false, false);
        // D[i,i]: pixel p<8 at (VGPR p, lane p); pixel p>=8 at (VGPR p-8, lane p+16)
        float dval = 0.f;
        #pragma unroll
        for (int r = 0; r < 8; ++r) {
            dval = (lane == r)      ? acc[r] : dval;
            dval = (lane == r + 24) ? acc[r] : dval;
        }
        if (lane < 8)   sc[o * 16 + lane]      = dval;
        if (lane >= 24) sc[o * 16 + lane - 16] = dval;
    }
    __syncthreads();

    // ---------------- stage 2: softmax over the 9 positions -----------------
    if (lane < 16) {
        float s[9], m = -1e30f;
        #pragma unroll
        for (int o = 0; o < 9; ++o) { s[o] = sc[o * 16 + lane] * SCALE; m = fmaxf(m, s[o]); }
        float sum = 0.f;
        #pragma unroll
        for (int o = 0; o < 9; ++o) { s[o] = __expf(s[o] - m); sum += s[o]; }
        const float inv = 1.f / sum;
        #pragma unroll
        for (int o = 0; o < 9; ++o) sc[o * 16 + lane] = s[o] * inv;
    }
    __syncthreads();

    // ---------------- stage 3: AV (fp32), write [B,H,W,C] -------------------
    // lane -> (pixel = xl, channel half db = 0 or 32)
    const int db = hi * 32;
    float wv[9];
    #pragma unroll
    for (int o = 0; o < 9; ++o) wv[o] = sc[o * 16 + xl];

    float4 acc[8];
    #pragma unroll
    for (int j = 0; j < 8; ++j) acc[j] = make_float4(0.f, 0.f, 0.f, 0.f);

    #pragma unroll
    for (int o = 0; o < 9; ++o) {
        const int dy = o / 3, dx = o % 3;
        const float4* vr = (const float4*)(vsm + (dy * 18 + xl + dx) * VS + db);
        const float w = wv[o];
        #pragma unroll
        for (int j = 0; j < 8; ++j) {
            const float4 t = vr[j];
            acc[j].x += w * t.x; acc[j].y += w * t.y;
            acc[j].z += w * t.z; acc[j].w += w * t.w;
        }
    }

    float4* op = (float4*)(out + (size_t)(((b * HH + y) * WW + x0 + xl) * CC + c0 + db));
    #pragma unroll
    for (int j = 0; j < 8; ++j) op[j] = acc[j];
}

extern "C" void kernel_launch(void* const* d_in, const int* in_sizes, int n_in,
                              void* d_out, int out_size, void* d_ws, size_t ws_size,
                              hipStream_t stream) {
    const float* q = (const float*)d_in[0];
    const float* k = (const float*)d_in[1];
    const float* v = (const float*)d_in[2];
    float* out = (float*)d_out;
    // one wave32 per (b, head, row, 16-pixel x-tile): 8*6*64*4 = 12288 blocks
    const int nblocks = BB * NH * HH * (WW / 16);
    dilate_attn_kernel<<<dim3(nblocks), dim3(32), 0, stream>>>(q, k, v, out);
}
